// Attention_2602750181655
// MI455X (gfx1250) — matrix operand order repaired and checked by reading, not compiled, not run on hardware
//
#include <hip/hip_runtime.h>

// ---------------- problem constants (from reference) ----------------
constexpr int B_   = 2;
constexpr int S_   = 2048;
constexpr int DIM_ = 2048;
constexpr int NH_  = 16;
constexpr int NKV_ = 4;
constexpr int HD_  = 128;          // head dim
constexpr int KVD_ = NKV_ * HD_;   // 512
constexpr int M_   = B_ * S_;      // 4096 rows for projections

typedef __attribute__((ext_vector_type(16))) __bf16 v16bf;
typedef __attribute__((ext_vector_type(8)))  __bf16 v8bf;
typedef __attribute__((ext_vector_type(8)))  float  v8f;

#define WMMA_BF16(a, b, c) \
    __builtin_amdgcn_wmma_f32_16x16x32_bf16(false, (a), false, (b), (short)0, (c), false, false)

static __device__ __forceinline__ v8f vzero8() {
    v8f z;
#pragma unroll
    for (int i = 0; i < 8; ++i) z[i] = 0.0f;
    return z;
}

static __device__ __forceinline__ v16bf cat8(v8bf lo, v8bf hi) {
    union { v16bf v; v8bf h[2]; } u;
    u.h[0] = lo; u.h[1] = hi;
    return u.v;
}

static __device__ __forceinline__ v8bf pack8(v8f v) {
    v8bf r;
#pragma unroll
    for (int i = 0; i < 8; ++i) r[i] = (__bf16)v[i];
    return r;
}

static __device__ __forceinline__ float vmax8(v8f v) {
    float a = fmaxf(fmaxf(v[0], v[1]), fmaxf(v[2], v[3]));
    float b = fmaxf(fmaxf(v[4], v[5]), fmaxf(v[6], v[7]));
    return fmaxf(a, b);
}
static __device__ __forceinline__ float vsum8(v8f v) {
    return ((v[0] + v[1]) + (v[2] + v[3])) + ((v[4] + v[5]) + (v[6] + v[7]));
}

// A fragment (16x32 bf16, M x K), A row-major with leading dim lda.
static __device__ __forceinline__ v16bf load_a_frag(const __bf16* A, int lda, int m0, int k0, int lane) {
    int row = m0 + (lane & 15);
    int off = (lane & 16) ? 16 : 0;
    const __bf16* p = A + (size_t)row * lda + k0 + off;
    v8bf lo = *(const v8bf*)p;
    v8bf hi = *(const v8bf*)(p + 8);
    return cat8(lo, hi);
}

// B fragment (32x16 bf16, K x N) supplied as Bt = B^T row-major [N x K], leading dim ldb.
static __device__ __forceinline__ v16bf load_b_frag(const __bf16* Bt, int ldb, int n0, int k0, int lane) {
    int col = n0 + (lane & 15);
    int off = (lane & 16) ? 16 : 0;
    const __bf16* p = Bt + (size_t)col * ldb + k0 + off;
    v8bf lo = *(const v8bf*)p;
    v8bf hi = *(const v8bf*)(p + 8);
    return cat8(lo, hi);
}

// ---------------- elementwise prep kernels ----------------
__global__ void cvt_f32_to_bf16(const float* __restrict__ in, __bf16* __restrict__ out, size_t n) {
    size_t i = (size_t)blockIdx.x * blockDim.x + threadIdx.x;
    if (i < n) out[i] = (__bf16)in[i];
}

// in: f32 [K x N] row-major  ->  out: bf16 [N x K] (transposed)
__global__ void cvt_f32_to_bf16_T(const float* __restrict__ in, __bf16* __restrict__ out, int K, int N) {
    size_t i = (size_t)blockIdx.x * blockDim.x + threadIdx.x;
    size_t total = (size_t)K * N;
    if (i >= total) return;
    int k = (int)(i % K);
    int n = (int)(i / K);
    out[(size_t)n * K + k] = (__bf16)in[(size_t)k * N + n];
}

// RoPE in place on bf16 tensor laid out [B, S, nheads, 128]
__global__ void rope_kernel(__bf16* __restrict__ t, const float* __restrict__ fc,
                            const float* __restrict__ fs, int nheads, size_t total) {
    size_t idx = (size_t)blockIdx.x * blockDim.x + threadIdx.x;
    if (idx >= total) return;
    int i = (int)(idx & 63);
    int h = (int)((idx >> 6) % nheads);
    int s = (int)((idx / ((size_t)64 * nheads)) % S_);
    int b = (int)(idx / ((size_t)64 * nheads * S_));
    float c  = fc[((size_t)b * S_ + s) * 64 + i];
    float sn = fs[((size_t)b * S_ + s) * 64 + i];
    size_t base = (((size_t)b * S_ + s) * nheads + h) * HD_ + 2 * i;
    float xr = (float)t[base];
    float xi = (float)t[base + 1];
    t[base]     = (__bf16)(xr * c - xi * sn);
    t[base + 1] = (__bf16)(xr * sn + xi * c);
}

// v: [B, S, 4, 128] -> vt: [B, 4, 128, S]
__global__ void transpose_v_kernel(const __bf16* __restrict__ v, __bf16* __restrict__ vt) {
    size_t i = (size_t)blockIdx.x * blockDim.x + threadIdx.x;
    size_t total = (size_t)B_ * NKV_ * HD_ * S_;
    if (i >= total) return;
    int s  = (int)(i % S_);
    int d  = (int)((i / S_) % HD_);
    int hk = (int)((i / ((size_t)S_ * HD_)) % NKV_);
    int b  = (int)(i / ((size_t)S_ * HD_ * NKV_));
    vt[i] = v[(((size_t)b * S_ + s) * NKV_ + hk) * HD_ + d];
}

// ---------------- bf16 WMMA GEMM: C[MxN] = A[MxK] @ Bt^T ----------------
// Each wave owns a 32x32 output macro-tile (2x2 WMMA tiles), software
// double-buffered: iteration k's WMMAs overlap iteration k+1's loads.
template <bool F32OUT>
__global__ void gemm_bf16_kernel(const __bf16* __restrict__ A, const __bf16* __restrict__ Bt,
                                 __bf16* __restrict__ Cb, float* __restrict__ Cf,
                                 int M, int N, int K) {
    int lane = threadIdx.x & 31;
    int wave = blockIdx.x * (blockDim.x >> 5) + (threadIdx.x >> 5);
    int mT = M >> 5, nT = N >> 5;
    if (wave >= mT * nT) return;
    int wm = (wave % mT) << 5;
    int wn = (wave / mT) << 5;

    v8f a00 = vzero8(), a01 = vzero8(), a10 = vzero8(), a11 = vzero8();

    v16bf fa0 = load_a_frag(A, K, wm,      0, lane);
    v16bf fa1 = load_a_frag(A, K, wm + 16, 0, lane);
    v16bf fb0 = load_b_frag(Bt, K, wn,      0, lane);
    v16bf fb1 = load_b_frag(Bt, K, wn + 16, 0, lane);

    for (int k0 = 0; k0 < K; k0 += 32) {
        int k1 = (k0 + 32 < K) ? k0 + 32 : k0;  // last iter: redundant reload
        v16bf na0 = load_a_frag(A, K, wm,      k1, lane);
        v16bf na1 = load_a_frag(A, K, wm + 16, k1, lane);
        v16bf nb0 = load_b_frag(Bt, K, wn,      k1, lane);
        v16bf nb1 = load_b_frag(Bt, K, wn + 16, k1, lane);
        if (k1 + 32 < K) {
            __builtin_prefetch(A  + (size_t)(wm + (lane & 15)) * K + k1 + 32, 0, 1);
            __builtin_prefetch(Bt + (size_t)(wn + (lane & 15)) * K + k1 + 32, 0, 1);
        }
        a00 = WMMA_BF16(fa0, fb0, a00);
        a01 = WMMA_BF16(fa0, fb1, a01);
        a10 = WMMA_BF16(fa1, fb0, a10);
        a11 = WMMA_BF16(fa1, fb1, a11);
        fa0 = na0; fa1 = na1; fb0 = nb0; fb1 = nb1;
    }

    int rOff = (lane & 16) ? 8 : 0;
    int cn   = lane & 15;
#pragma unroll
    for (int r = 0; r < 8; ++r) {
        int m0 = wm + r + rOff;
        int m1 = wm + 16 + r + rOff;
        if (F32OUT) {
            Cf[(size_t)m0 * N + wn + cn]      = a00[r];
            Cf[(size_t)m0 * N + wn + 16 + cn] = a01[r];
            Cf[(size_t)m1 * N + wn + cn]      = a10[r];
            Cf[(size_t)m1 * N + wn + 16 + cn] = a11[r];
        } else {
            Cb[(size_t)m0 * N + wn + cn]      = (__bf16)a00[r];
            Cb[(size_t)m0 * N + wn + 16 + cn] = (__bf16)a01[r];
            Cb[(size_t)m1 * N + wn + cn]      = (__bf16)a10[r];
            Cb[(size_t)m1 * N + wn + 16 + cn] = (__bf16)a11[r];
        }
    }
}

// ---------------- flash attention (transposed-score formulation) ----------------
// Per 16-query tile:  S^T = K_chunk(32x128) @ Q^T(128x16)  (two 16x16 C tiles)
// Each lane owns ONE query column; softmax stats are per-lane scalars merged
// with a single shfl_xor(16).  O accumulated transposed: O^T = V^T @ P^T.
// All 8 K fragments are loaded up front (one clause) so score WMMAs overlap
// the loads; V fragment loads are issued BEFORE the softmax so the softmax
// hides their latency.
__global__ __launch_bounds__(128, 1)
void flash_attn_kernel(const __bf16* __restrict__ q, const __bf16* __restrict__ k,
                       const __bf16* __restrict__ vt, __bf16* __restrict__ ao) {
    constexpr int PLD = 40;  // LDS row stride (bf16): conflict-free, 16B aligned
    __shared__ __align__(16) __bf16 pbuf[4][16 * PLD];

    int lane = threadIdx.x & 31;
    int wvid = threadIdx.x >> 5;
    int tile = blockIdx.x * 4 + wvid;    // B*NH*(S/16) = 4096 tiles, grid exact

    const int QT = S_ / 16;              // 128
    int qt = tile % QT;
    int h  = (tile / QT) % NH_;
    int b  = tile / (QT * NH_);
    int hkv = h / (NH_ / NKV_);

    const __bf16* qh  = q  + (size_t)b * S_ * DIM_ + (size_t)h * HD_;      // ldb = DIM_
    const __bf16* kh  = k  + (size_t)b * S_ * KVD_ + (size_t)hkv * HD_;    // lda = KVD_
    const __bf16* vth = vt + ((size_t)b * NKV_ + hkv) * HD_ * S_;          // lda = S_

    int hi8 = (lane & 16) ? 8 : 0;
    int cq  = lane & 15;
    int qi  = qt * 16 + cq;              // this lane's query row

    // preload Q^T B-fragments (K dim = 128 -> 4 fragments)
    v16bf bq[4];
#pragma unroll
    for (int kc = 0; kc < 4; ++kc) bq[kc] = load_b_frag(qh, DIM_, qt * 16, kc * 32, lane);

    v8f o[8];
#pragma unroll
    for (int t = 0; t < 8; ++t) o[t] = vzero8();
    float m = -3.0e38f, l = 0.0f;

    const float alpha = 0.08838834764831845f * 1.4426950408889634f; // 1/sqrt(128)*log2(e)
    int nch = ((qt + 1) * 16 + 31) >> 5;

    for (int c = 0; c < nch; ++c) {
        int kvb = c * 32;

        if (c + 1 < nch) {
            __builtin_prefetch(kh + (size_t)(kvb + 32 + cq) * KVD_, 0, 1);
            __builtin_prefetch(vth + (size_t)cq * S_ + kvb + 32, 0, 1);
        }

        // ---- load ALL K fragments for this chunk first (loads overlap WMMAs) ----
        v16bf ka[8];
#pragma unroll
        for (int kc = 0; kc < 4; ++kc) {
            ka[2 * kc]     = load_a_frag(kh, KVD_, kvb,      kc * 32, lane);
            ka[2 * kc + 1] = load_a_frag(kh, KVD_, kvb + 16, kc * 32, lane);
        }

        // ---- S^T tiles: (kv rows) x (q cols) ----
        v8f s0 = vzero8(), s1 = vzero8();
#pragma unroll
        for (int kc = 0; kc < 4; ++kc) {
            s0 = WMMA_BF16(ka[2 * kc],     bq[kc], s0);
            s1 = WMMA_BF16(ka[2 * kc + 1], bq[kc], s1);
        }

        // ---- issue V fragment loads NOW; softmax below hides their latency ----
        v16bf va[8];
#pragma unroll
        for (int t = 0; t < 8; ++t) va[t] = load_a_frag(vth, S_, 16 * t, kvb, lane);

        // ---- causal mask + scale (element (r,lane): kv = kvb + 16t + r + hi8) ----
        v8f f0, f1;
#pragma unroll
        for (int r = 0; r < 8; ++r) {
            f0[r] = (kvb + r + hi8      <= qi) ? s0[r] * alpha : -1.0e30f;
            f1[r] = (kvb + 16 + r + hi8 <= qi) ? s1[r] * alpha : -1.0e30f;
        }

        // ---- per-lane online softmax ----
        float cm = fmaxf(vmax8(f0), vmax8(f1));
        cm = fmaxf(cm, __shfl_xor(cm, 16, 32));
        float mn  = fmaxf(m, cm);
        float rsc = exp2f(m - mn);
        m = mn;

        v8f p0, p1;
#pragma unroll
        for (int r = 0; r < 8; ++r) {
            p0[r] = exp2f(f0[r] - mn);
            p1[r] = exp2f(f1[r] - mn);
        }
        float rs = vsum8(p0) + vsum8(p1);
        rs += __shfl_xor(rs, 16, 32);
        l = l * rsc + rs;

#pragma unroll
        for (int t = 0; t < 8; ++t) o[t] = o[t] * rsc;

        // ---- stage P as [q][kv] in LDS (packed 16B stores) ----
        *(v8bf*)&pbuf[wvid][cq * PLD + hi8]      = pack8(p0);
        *(v8bf*)&pbuf[wvid][cq * PLD + 16 + hi8] = pack8(p1);
        asm volatile("s_wait_dscnt 0" ::: "memory");

        // ---- P^T B-fragment: lane col q = cq, kv run [0,16) / [16,32) ----
        const __bf16* pp = &pbuf[wvid][cq * PLD + ((lane & 16) ? 16 : 0)];
        v16bf pb = cat8(*(const v8bf*)pp, *(const v8bf*)(pp + 8));

        // ---- O^T += V^T(16d x 32kv) @ P^T(32kv x 16q) ----
#pragma unroll
        for (int t = 0; t < 8; ++t) o[t] = WMMA_BF16(va[t], pb, o[t]);

        asm volatile("" ::: "memory");
    }

    // ---- normalize, pack, store (O^T element (t,r): d = 16t + r + hi8, q = cq) ----
    float il = 1.0f / l;
    __bf16* aor = ao + ((size_t)b * S_ + qi) * DIM_ + (size_t)h * HD_;
#pragma unroll
    for (int t = 0; t < 8; ++t) {
        v8bf pk = pack8(o[t] * il);
        *(v8bf*)(aor + 16 * t + hi8) = pk;
    }
}

// ---------------- host-side orchestration ----------------
extern "C" void kernel_launch(void* const* d_in, const int* in_sizes, int n_in,
                              void* d_out, int out_size, void* d_ws, size_t ws_size,
                              hipStream_t stream) {
    (void)in_sizes; (void)n_in; (void)out_size; (void)ws_size;

    const float* x  = (const float*)d_in[0];
    const float* wq = (const float*)d_in[1];
    const float* wk = (const float*)d_in[2];
    const float* wv = (const float*)d_in[3];
    const float* wo = (const float*)d_in[4];
    const float* fc = (const float*)d_in[5];
    const float* fs = (const float*)d_in[6];
    float* out = (float*)d_out;

    char* ws = (char*)d_ws;
    const size_t SZ_XB  = (size_t)M_ * DIM_ * 2;        // 16 MB
    const size_t SZ_WQT = (size_t)DIM_ * DIM_ * 2;      //  8 MB
    const size_t SZ_WKT = (size_t)KVD_ * DIM_ * 2;      //  2 MB
    const size_t SZ_Q   = (size_t)M_ * DIM_ * 2;        // 16 MB
    const size_t SZ_K   = (size_t)M_ * KVD_ * 2;        //  4 MB

    size_t off = 0;
    __bf16* xb  = (__bf16*)(ws + off); off += SZ_XB;
    __bf16* wqt = (__bf16*)(ws + off); off += SZ_WQT;
    __bf16* wkt = (__bf16*)(ws + off); off += SZ_WKT;
    __bf16* wvt = (__bf16*)(ws + off); off += SZ_WKT;
    __bf16* wot = (__bf16*)(ws + off); off += SZ_WQT;
    __bf16* qb  = (__bf16*)(ws + off); off += SZ_Q;
    __bf16* kb  = (__bf16*)(ws + off); off += SZ_K;
    __bf16* vb  = (__bf16*)(ws + off); off += SZ_K;
    __bf16* vtb = (__bf16*)(ws + off); off += SZ_K;
    __bf16* aob = (__bf16*)(ws + off); off += SZ_Q;     // total ~80 MB

    const int TB = 256;

    // 1) convert x to bf16; transpose+convert weights to (N, K) bf16
    {
        size_t n = (size_t)M_ * DIM_;
        cvt_f32_to_bf16<<<(unsigned)((n + TB - 1) / TB), TB, 0, stream>>>(x, xb, n);
        size_t nw = (size_t)DIM_ * DIM_;
        cvt_f32_to_bf16_T<<<(unsigned)((nw + TB - 1) / TB), TB, 0, stream>>>(wq, wqt, DIM_, DIM_);
        size_t nk = (size_t)DIM_ * KVD_;
        cvt_f32_to_bf16_T<<<(unsigned)((nk + TB - 1) / TB), TB, 0, stream>>>(wk, wkt, DIM_, KVD_);
        cvt_f32_to_bf16_T<<<(unsigned)((nk + TB - 1) / TB), TB, 0, stream>>>(wv, wvt, DIM_, KVD_);
        cvt_f32_to_bf16_T<<<(unsigned)((nw + TB - 1) / TB), TB, 0, stream>>>(wo, wot, DIM_, DIM_);
    }

    // 2) QKV projections (bf16 WMMA GEMM)
    {
        int waves = (M_ >> 5) * (DIM_ >> 5);   // 8192
        gemm_bf16_kernel<false><<<waves / 4, 128, 0, stream>>>(xb, wqt, qb, nullptr, M_, DIM_, DIM_);
        waves = (M_ >> 5) * (KVD_ >> 5);       // 2048
        gemm_bf16_kernel<false><<<waves / 4, 128, 0, stream>>>(xb, wkt, kb, nullptr, M_, KVD_, DIM_);
        gemm_bf16_kernel<false><<<waves / 4, 128, 0, stream>>>(xb, wvt, vb, nullptr, M_, KVD_, DIM_);
    }

    // 3) RoPE on Q and K, transpose V per kv-head
    {
        size_t nq = (size_t)B_ * S_ * NH_ * 64;
        rope_kernel<<<(unsigned)((nq + TB - 1) / TB), TB, 0, stream>>>(qb, fc, fs, NH_, nq);
        size_t nk = (size_t)B_ * S_ * NKV_ * 64;
        rope_kernel<<<(unsigned)((nk + TB - 1) / TB), TB, 0, stream>>>(kb, fc, fs, NKV_, nk);
        size_t nv = (size_t)B_ * NKV_ * HD_ * S_;
        transpose_v_kernel<<<(unsigned)((nv + TB - 1) / TB), TB, 0, stream>>>(vb, vtb);
    }

    // 4) flash attention: 4096 query tiles, 4 waves per block
    {
        int tiles = B_ * NH_ * (S_ / 16);      // 4096
        flash_attn_kernel<<<tiles / 4, 128, 0, stream>>>(qb, kb, vtb, aob);
    }

    // 5) output projection -> fp32 d_out
    {
        int waves = (M_ >> 5) * (DIM_ >> 5);   // 8192
        gemm_bf16_kernel<true><<<waves / 4, 128, 0, stream>>>(aob, wot, nullptr, out, M_, DIM_, DIM_);
    }
}
